// MSTransformerDecoder_6880537608754
// MI455X (gfx1250) — 8-bit matrix operand order repaired and checked by reading, compiled, not run on hardware
//
#include <hip/hip_runtime.h>


#define TBSZ   8
#define NN     1024
#define CC     256
#define HH     1024
#define NHEAD  8
#define DH     32
#define SCALE_F 0.1f
#define KPAD   40

typedef __attribute__((ext_vector_type(16))) _Float16 v16h;
typedef __attribute__((ext_vector_type(8)))  float    v8f;
typedef __attribute__((ext_vector_type(8)))  int      v8i;

__device__ __forceinline__ float spike_rnd(float x) {
    float r = __builtin_rintf(x);
    return fminf(fmaxf(r, 0.f), 4.f);
}

__global__ void wcvt_kernel(const float* __restrict__ src,
                            _Float16* __restrict__ dst, int n) {
    int i = blockIdx.x * blockDim.x + threadIdx.x;
    if (i < n) dst[i] = (_Float16)src[i];
}

template <int INMODE, int EPI>
__global__ __launch_bounds__(256) void conv_gemm_kernel(
    const void* __restrict__ inp, const _Float16* __restrict__ W,
    const float* __restrict__ bb, const float* __restrict__ ss,
    const float* __restrict__ tt, float inv,
    unsigned char* __restrict__ outU8, float* __restrict__ outF,
    const float* __restrict__ resid, int M, int K) {
    __shared__ __align__(16) _Float16 Bl[128 * KPAD];

    const int tb   = blockIdx.z;
    const int nblk = blockIdx.x * 128;
    const int mblk = blockIdx.y * 128;
    const int t    = threadIdx.x;
    const int lane = t & 31;
    const int wave = t >> 5;
    const int wm   = (wave & 3) * 32;
    const int wn   = (wave >> 2) * 64;

    const unsigned char* in8 = (const unsigned char*)inp + (size_t)tb * K * NN;
    const float*         inf = (const float*)inp + (size_t)tb * K * NN;

    v8f acc[2][4];
#pragma unroll
    for (int i = 0; i < 2; ++i)
#pragma unroll
        for (int j = 0; j < 4; ++j)
#pragma unroll
            for (int r = 0; r < 8; ++r) acc[i][j][r] = 0.f;

    const int k4 = (t & 7) * 4;
    const int n4 = (t >> 3) * 4;

    for (int kk = 0; kk < K; kk += 32) {
        {
            _Float16 hv[4][4];
#pragma unroll
            for (int kq = 0; kq < 4; ++kq) {
                const size_t rowoff = (size_t)(kk + k4 + kq) * NN + nblk + n4;
                if (INMODE == 0) {
                    uchar4 u = *(const uchar4*)(in8 + rowoff);
                    hv[kq][0] = (_Float16)(float)u.x; hv[kq][1] = (_Float16)(float)u.y;
                    hv[kq][2] = (_Float16)(float)u.z; hv[kq][3] = (_Float16)(float)u.w;
                } else {
                    float4 f = *(const float4*)(inf + rowoff);
                    float a0 = spike_rnd(f.x), a1 = spike_rnd(f.y);
                    float a2 = spike_rnd(f.z), a3 = spike_rnd(f.w);
                    if (INMODE == 2) {
                        a0 = (a0 > 2.f) ? 1.f : 0.f; a1 = (a1 > 2.f) ? 1.f : 0.f;
                        a2 = (a2 > 2.f) ? 1.f : 0.f; a3 = (a3 > 2.f) ? 1.f : 0.f;
                    }
                    hv[kq][0] = (_Float16)a0; hv[kq][1] = (_Float16)a1;
                    hv[kq][2] = (_Float16)a2; hv[kq][3] = (_Float16)a3;
                }
            }
#pragma unroll
            for (int nn = 0; nn < 4; ++nn) {
                _Float16 tmp[4] = {hv[0][nn], hv[1][nn], hv[2][nn], hv[3][nn]};
                *(uint2*)&Bl[(n4 + nn) * KPAD + k4] = *(const uint2*)tmp;
            }
        }
        __syncthreads();

        v16h af[2];
#pragma unroll
        for (int i = 0; i < 2; ++i) {
            const int row = mblk + wm + i * 16 + (lane & 15);
            const int kh  = kk + ((lane >> 4) << 3);
            const _Float16* p = W + (size_t)row * K + kh;
            uint4* ap = (uint4*)&af[i];
            ap[0] = *(const uint4*)p;
            ap[1] = *(const uint4*)(p + 16);
        }
        v16h bf[4];
#pragma unroll
        for (int j = 0; j < 4; ++j) {
            const _Float16* p =
                &Bl[(wn + j * 16 + (lane & 15)) * KPAD + ((lane >> 4) << 3)];
            uint4* bp = (uint4*)&bf[j];
            bp[0] = *(const uint4*)p;
            bp[1] = *(const uint4*)(p + 16);
        }
#pragma unroll
        for (int i = 0; i < 2; ++i)
#pragma unroll
            for (int j = 0; j < 4; ++j)
                acc[i][j] = __builtin_amdgcn_wmma_f32_16x16x32_f16(
                    false, af[i], false, bf[j], (short)0, acc[i][j], false, false);
        __syncthreads();
    }

#pragma unroll
    for (int i = 0; i < 2; ++i) {
#pragma unroll
        for (int r = 0; r < 8; ++r) {
            const int o = mblk + wm + i * 16 + ((lane >> 4) << 3) + r;
            const float sv = ss[o];
            const float cb = bb[o] * sv + tt[o];
            const float cs = inv * sv;
#pragma unroll
            for (int j = 0; j < 4; ++j) {
                const int n = nblk + wn + j * 16 + (lane & 15);
                const float y = acc[i][j][r] * cs + cb;
                const size_t idx = (size_t)tb * M * NN + (size_t)o * NN + n;
                if (EPI == 0) outU8[idx] = (unsigned char)spike_rnd(y);
                else          outF[idx]  = y + 2.f * resid[idx];
            }
        }
    }
}

__global__ __launch_bounds__(256) void attn_kernel(
    const unsigned char* __restrict__ Q, const unsigned char* __restrict__ Kc,
    const unsigned char* __restrict__ Vv, unsigned char* __restrict__ O) {
    __shared__ __align__(16) unsigned char qs[128 * 64];
    __shared__ __align__(16) unsigned char ks[64 * 64];
    __shared__ __align__(16) _Float16      vs[32 * 72];

    const int tb   = blockIdx.z;
    const int h    = blockIdx.y;
    const int nblk = blockIdx.x * 128;
    const int t    = threadIdx.x;
    const int lane = t & 31;
    const int wave = t >> 5;

    const size_t base = (size_t)tb * CC * NN + (size_t)h * DH * NN;
    const unsigned char* qp = Q + base;
    const unsigned char* kp = Kc + base;
    const unsigned char* vp = Vv + base;

    {
        const int d4 = (t & 7) * 4;
        const int n4 = (t >> 3) * 4;
        uchar4 u[4];
#pragma unroll
        for (int dq = 0; dq < 4; ++dq)
            u[dq] = *(const uchar4*)(qp + (size_t)(d4 + dq) * NN + nblk + n4);
        unsigned char col[4][4];
#pragma unroll
        for (int dq = 0; dq < 4; ++dq) {
            col[0][dq] = u[dq].x; col[1][dq] = u[dq].y;
            col[2][dq] = u[dq].z; col[3][dq] = u[dq].w;
        }
#pragma unroll
        for (int nn = 0; nn < 4; ++nn)
            *(unsigned int*)&qs[(n4 + nn) * 64 + d4] = *(const unsigned int*)col[nn];
        const int zr = t >> 1, zo = 32 + (t & 1) * 16;
        *(int4*)&qs[zr * 64 + zo] = make_int4(0, 0, 0, 0);
    }
    __syncthreads();

    v8i qb;
    {
        const int ncol = wave * 16 + (lane & 15);
        const int kb   = (lane >> 4) * 16;
        uint4* bp = (uint4*)&qb;
        bp[0] = *(const uint4*)&qs[ncol * 64 + kb];
        bp[1] = *(const uint4*)&qs[ncol * 64 + 32 + kb];
    }
    v8i zi;
#pragma unroll
    for (int r = 0; r < 8; ++r) zi[r] = 0;

    v8f oacc[2];
#pragma unroll
    for (int r = 0; r < 8; ++r) { oacc[0][r] = 0.f; oacc[1][r] = 0.f; }

    for (int m0 = 0; m0 < NN; m0 += 64) {
        __syncthreads();
        if (t < 128) {
            const int d4 = (t & 7) * 4;
            const int m4 = (t >> 3) * 4;
            uchar4 u[4];
#pragma unroll
            for (int dq = 0; dq < 4; ++dq)
                u[dq] = *(const uchar4*)(kp + (size_t)(d4 + dq) * NN + m0 + m4);
            unsigned char col[4][4];
#pragma unroll
            for (int dq = 0; dq < 4; ++dq) {
                col[0][dq] = u[dq].x; col[1][dq] = u[dq].y;
                col[2][dq] = u[dq].z; col[3][dq] = u[dq].w;
            }
#pragma unroll
            for (int mm = 0; mm < 4; ++mm)
                *(unsigned int*)&ks[(m4 + mm) * 64 + d4] = *(const unsigned int*)col[mm];
        } else {
            const int t2 = t - 128;
            const int zr = t2 >> 1, zo = 32 + (t2 & 1) * 16;
            *(int4*)&ks[zr * 64 + zo] = make_int4(0, 0, 0, 0);
        }
        {
            const int d  = t >> 3;
            const int mg = (t & 7) * 8;
#pragma unroll
            for (int j = 0; j < 8; j += 4) {
                uchar4 u = *(const uchar4*)(vp + (size_t)d * NN + m0 + mg + j);
                _Float16 tmp[4] = {(_Float16)(float)u.x, (_Float16)(float)u.y,
                                   (_Float16)(float)u.z, (_Float16)(float)u.w};
                *(uint2*)&vs[d * 72 + mg + j] = *(const uint2*)tmp;
            }
        }
        __syncthreads();

        v8i att[4];
#pragma unroll
        for (int j = 0; j < 4; ++j) {
            v8i ak;
            const int row = j * 16 + (lane & 15);
            const int kb  = (lane >> 4) * 16;
            uint4* ap = (uint4*)&ak;
            ap[0] = *(const uint4*)&ks[row * 64 + kb];
            ap[1] = *(const uint4*)&ks[row * 64 + 32 + kb];
            att[j] = __builtin_amdgcn_wmma_i32_16x16x64_iu8(
                false, ak, false, qb, zi, false, false);
        }

#pragma unroll
        for (int kb2 = 0; kb2 < 64; kb2 += 32) {
            const int j0 = kb2 >> 4;
            v16h bt;
#pragma unroll
            for (int i = 0; i < 8; ++i) {
                bt[i]     = (_Float16)(float)att[j0][i];
                bt[8 + i] = (_Float16)(float)att[j0 + 1][i];
            }
#pragma unroll
            for (int i2 = 0; i2 < 2; ++i2) {
                v16h av;
                const int drow = i2 * 16 + (lane & 15);
                const int kh   = (lane >> 4) * 8;
                uint4* ap = (uint4*)&av;
                ap[0] = *(const uint4*)&vs[drow * 72 + kb2 + kh];
                ap[1] = *(const uint4*)&vs[drow * 72 + kb2 + 16 + kh];
                oacc[i2] = __builtin_amdgcn_wmma_f32_16x16x32_f16(
                    false, av, false, bt, (short)0, oacc[i2], false, false);
            }
        }
    }

    const float fs = SCALE_F / 64.f;
#pragma unroll
    for (int i2 = 0; i2 < 2; ++i2) {
        const int n = nblk + wave * 16 + (lane & 15);
#pragma unroll
        for (int r = 0; r < 8; ++r) {
            const int d = i2 * 16 + (lane >> 4) * 8 + r;
            const float y = oacc[i2][r] * fs;
            O[base + (size_t)d * NN + n] = (unsigned char)spike_rnd(y);
        }
    }
}

extern "C" void kernel_launch(void* const* d_in, const int* in_sizes, int n_in,
                              void* d_out, int out_size, void* d_ws, size_t ws_size,
                              hipStream_t stream) {
    (void)in_sizes; (void)n_in; (void)out_size; (void)ws_size;
    const float* x  = (const float*)d_in[0];
    const float* Wa = (const float*)d_in[1];
    const float* ba = (const float*)d_in[2];
    const float* sa = (const float*)d_in[3];
    const float* ta = (const float*)d_in[4];
    const float* w1 = (const float*)d_in[5];
    const float* b1 = (const float*)d_in[6];
    const float* s1 = (const float*)d_in[7];
    const float* t1 = (const float*)d_in[8];
    const float* w2 = (const float*)d_in[9];
    const float* b2 = (const float*)d_in[10];
    const float* s2 = (const float*)d_in[11];
    const float* t2 = (const float*)d_in[12];
    float* out = (float*)d_out;

    char* ws = (char*)d_ws;
    size_t off = 0;
    _Float16* W16 = (_Float16*)(ws + off); off += (size_t)(8 * CC * CC + HH * CC + CC * HH) * 2;
    off = (off + 255) & ~(size_t)255;
    unsigned char* qbuf = (unsigned char*)(ws + off); off += (size_t)TBSZ * CC * NN;
    unsigned char* kbuf = (unsigned char*)(ws + off); off += (size_t)TBSZ * CC * NN;
    unsigned char* vbuf = (unsigned char*)(ws + off); off += (size_t)TBSZ * CC * NN;
    unsigned char* abuf = (unsigned char*)(ws + off); off += (size_t)TBSZ * CC * NN;
    unsigned char* hid  = (unsigned char*)(ws + off); off += (size_t)TBSZ * HH * NN;
    off = (off + 255) & ~(size_t)255;
    float* x1 = (float*)(ws + off); off += (size_t)TBSZ * NN * CC * 4;
    float* x2 = (float*)(ws + off); off += (size_t)TBSZ * NN * CC * 4;

    _Float16* WA16 = W16;
    _Float16* W116 = W16 + (size_t)8 * CC * CC;
    _Float16* W216 = W116 + (size_t)HH * CC;

    const int nWa = 8 * CC * CC, n1 = HH * CC, n2 = CC * HH;
    wcvt_kernel<<<(nWa + 255) / 256, 256, 0, stream>>>(Wa, WA16, nWa);
    wcvt_kernel<<<(n1 + 255) / 256, 256, 0, stream>>>(w1, W116, n1);
    wcvt_kernel<<<(n2 + 255) / 256, 256, 0, stream>>>(w2, W216, n2);

    dim3 blk(256);
    dim3 gc(NN / 128, CC / 128, TBSZ);
    dim3 gh(NN / 128, HH / 128, TBSZ);
    dim3 ga(NN / 128, NHEAD, TBSZ);
    auto Wm = [&](int i) { return WA16 + (size_t)i * CC * CC; };

    conv_gemm_kernel<2, 0><<<gc, blk, 0, stream>>>(x, Wm(0), ba + 0 * CC, sa + 0 * CC, ta + 0 * CC,
                                                   1.00f, qbuf, nullptr, nullptr, CC, CC);
    conv_gemm_kernel<1, 0><<<gc, blk, 0, stream>>>(x, Wm(1), ba + 1 * CC, sa + 1 * CC, ta + 1 * CC,
                                                   0.25f, kbuf, nullptr, nullptr, CC, CC);
    conv_gemm_kernel<1, 0><<<gc, blk, 0, stream>>>(x, Wm(2), ba + 2 * CC, sa + 2 * CC, ta + 2 * CC,
                                                   0.25f, vbuf, nullptr, nullptr, CC, CC);
    attn_kernel<<<ga, blk, 0, stream>>>(qbuf, kbuf, vbuf, abuf);
    conv_gemm_kernel<0, 1><<<gc, blk, 0, stream>>>(abuf, Wm(3), ba + 3 * CC, sa + 3 * CC, ta + 3 * CC,
                                                   0.25f, nullptr, x1, x, CC, CC);

    conv_gemm_kernel<1, 0><<<gc, blk, 0, stream>>>(x1, Wm(4), ba + 4 * CC, sa + 4 * CC, ta + 4 * CC,
                                                   0.25f, qbuf, nullptr, nullptr, CC, CC);
    conv_gemm_kernel<1, 0><<<gc, blk, 0, stream>>>(x1, Wm(5), ba + 5 * CC, sa + 5 * CC, ta + 5 * CC,
                                                   0.25f, kbuf, nullptr, nullptr, CC, CC);
    conv_gemm_kernel<1, 0><<<gc, blk, 0, stream>>>(x1, Wm(6), ba + 6 * CC, sa + 6 * CC, ta + 6 * CC,
                                                   0.25f, vbuf, nullptr, nullptr, CC, CC);
    attn_kernel<<<ga, blk, 0, stream>>>(qbuf, kbuf, vbuf, abuf);
    conv_gemm_kernel<0, 1><<<gc, blk, 0, stream>>>(abuf, Wm(7), ba + 7 * CC, sa + 7 * CC, ta + 7 * CC,
                                                   0.25f, nullptr, x2, x1, CC, CC);

    conv_gemm_kernel<1, 0><<<gh, blk, 0, stream>>>(x2, W116, b1, s1, t1,
                                                   0.25f, hid, nullptr, nullptr, HH, CC);
    conv_gemm_kernel<0, 1><<<gc, blk, 0, stream>>>(hid, W216, b2, s2, t2,
                                                   0.25f, nullptr, out, x2, CC, HH);
}
